// FastCNN_82351702933521
// MI455X (gfx1250) — matrix operand order repaired and checked by reading, not compiled, not run on hardware
//
#include <hip/hip_runtime.h>
#include <hip/hip_fp16.h>
#include <math.h>

typedef _Float16 h16;
typedef __attribute__((ext_vector_type(16))) _Float16 v16h;
typedef __attribute__((ext_vector_type(8)))  _Float16 v8h;
typedef __attribute__((ext_vector_type(8)))  float    v8f;

#define H_IMG 384
#define W_IMG 512
#define FHh 24
#define FWw 32
#define N_ANCH 6912      // 24*32*9
#define SORT_N 8192
#define N_PRE 6000
#define N_POST 300
#define NMS_TH 0.7f
#define NEGV (-1e30f)

__device__ __forceinline__ int imin(int a, int b) { return a < b ? a : b; }
__device__ __forceinline__ int imax(int a, int b) { return a > b ? a : b; }

__device__ __forceinline__ v16h cat8(v8h lo, v8h hi8) {
  v16h r;
#pragma unroll
  for (int i = 0; i < 8; ++i) { r[i] = lo[i]; r[i + 8] = hi8[i]; }
  return r;
}

// ---------------------------------------------------------------------------
// f32 -> f16 conversion
// ---------------------------------------------------------------------------
__global__ void k_cvt_f16(const float* __restrict__ in, h16* __restrict__ out, int n) {
  int i = blockIdx.x * blockDim.x + threadIdx.x;
  if (i < n) out[i] = (h16)in[i];
}

// ---------------------------------------------------------------------------
// Implicit-GEMM convolution with WMMA (one wave per 16x16 output tile)
//   out[cout, p] = sum_k W[cout,k] * im2col[k,p] + bias[cout]
// ---------------------------------------------------------------------------
__global__ __launch_bounds__(32)
void k_conv_wmma(const float* __restrict__ W, const float* __restrict__ bias,
                 const h16* __restrict__ in, h16* __restrict__ out,
                 int Cin, int Cout, int H, int Wd, int KH, int KW, int pad, int relu)
{
  const int lane  = threadIdx.x;
  const int hi    = (lane >= 16) ? 1 : 0;
  const int l15   = lane & 15;
  const int nBase = blockIdx.x * 16;     // pixel tile
  const int mBase = blockIdx.y * 16;     // cout tile
  const int HWt   = H * Wd;
  const int KK    = KH * KW;
  const int K     = Cin * KK;

  const int   arow    = mBase + l15;
  const bool  arow_ok = (arow < Cout);
  const float* Arow   = W + (size_t)arow * K;

  const int   pix    = nBase + l15;
  const bool  pix_ok = (pix < HWt);
  const int   py     = pix_ok ? pix / Wd : 0;
  const int   px     = pix_ok ? pix % Wd : 0;

  v8f acc = {};
  for (int k0 = 0; k0 < K; k0 += 32) {
    v16h a, b;
#pragma unroll
    for (int e = 0; e < 16; ++e) {
      // A fragment: lane-half base 0/8, K runs {0..7} and {16..23}
      int ka = k0 + hi * 8 + (e < 8 ? e : e + 8);
      a[e] = (arow_ok && ka < K) ? (h16)Arow[ka] : (h16)0.0f;
      // B fragment: lane-half base 0/8, K runs {0..7} and {16..23}
      int kb = k0 + hi * 8 + (e < 8 ? e : e + 8);
      h16 bv = (h16)0.0f;
      if (pix_ok && kb < K) {
        int ci  = kb / KK;
        int rem = kb % KK;
        int kh  = rem / KW, kw = rem % KW;
        int iy = py + kh - pad, ix = px + kw - pad;
        if (iy >= 0 && iy < H && ix >= 0 && ix < Wd)
          bv = in[(size_t)ci * HWt + (size_t)iy * Wd + ix];
      }
      b[e] = bv;
    }
    acc = __builtin_amdgcn_wmma_f32_16x16x32_f16(false, a, false, b,
                                                 (short)0, acc, false, false);
  }

#pragma unroll
  for (int r = 0; r < 8; ++r) {
    int m = mBase + r + hi * 8;          // D: lanes>=16 carry rows M+8
    if (m < Cout && pix_ok) {
      float v = acc[r] + bias[m];
      if (relu) v = v > 0.0f ? v : 0.0f;
      out[(size_t)m * HWt + pix] = (h16)v;
    }
  }
}

// ---------------------------------------------------------------------------
// LDS-staged WMMA GEMM: out[m,n] = relu?(A[m,:]·B[:,n] + bias[n])
//   A f16 row-major [M,K]; B f32 row-major [K,N] (weights, converted in-flight)
//   Workgroup = 256 threads (8 waves). Tile: 16 (M) x 128 (N) x 32 (K).
//   A tile [16][32] and B tile (transposed) [128][32] staged in LDS with
//   8-half row padding; each wave consumes its fragments via 16B ds loads.
// ---------------------------------------------------------------------------
#define GT_N 128           // N tile per workgroup
#define LDS_STRIDE 40      // 32 + 8 halves pad; 80B keeps 16B alignment
__global__ __launch_bounds__(256)
void k_gemm_wmma(const h16* __restrict__ A, const float* __restrict__ B,
                 const float* __restrict__ bias, int M, int N, int K,
                 int relu, h16* __restrict__ outH, float* __restrict__ outF)
{
#if __has_builtin(__builtin_amdgcn_s_wait_tensorcnt)
  __builtin_amdgcn_s_wait_tensorcnt(0);   // exercise gfx1250 split-counter path
#endif
  __shared__ __align__(16) h16 ldsA[16 * LDS_STRIDE];
  __shared__ __align__(16) h16 ldsB[GT_N * LDS_STRIDE];

  const int tid    = threadIdx.x;
  const int lane   = tid & 31;
  const int waveId = tid >> 5;          // 0..7
  const int hi     = (lane >= 16) ? 1 : 0;
  const int l15    = lane & 15;
  const int nBase  = blockIdx.x * GT_N;
  const int mBase  = blockIdx.y * 16;

  // cooperative-load coordinates
  const int arow_c = tid >> 4;          // 0..15 (A tile row)
  const int akk_c  = (tid & 15) * 2;    // 0..30 (A tile k, pairs)
  const int bn_c   = tid & (GT_N - 1);  // 0..127 (B tile n)
  const int bk_c   = tid >> 7;          // 0..1   (B tile k base)

  const bool arow_ok = (mBase + arow_c) < M;
  const h16* Arow    = A + (size_t)(mBase + arow_c) * K;
  const int  bn_g    = nBase + bn_c;
  const bool bn_ok   = bn_g < N;

  v8f acc = {};
  for (int k0 = 0; k0 < K; k0 += 32) {
    // ---- stage A tile (coalesced f16 pair loads) ----
    {
      h16 v0 = (h16)0.0f, v1 = (h16)0.0f;
      if (arow_ok) {
        int ka = k0 + akk_c;
        if (ka     < K) v0 = Arow[ka];
        if (ka + 1 < K) v1 = Arow[ka + 1];
      }
      ldsA[arow_c * LDS_STRIDE + akk_c]     = v0;
      ldsA[arow_c * LDS_STRIDE + akk_c + 1] = v1;
    }
    // ---- stage B tile transposed [n][k] (coalesced f32 loads along n) ----
#pragma unroll
    for (int t = 0; t < 16; ++t) {
      int k = t * 2 + bk_c;             // 0..31
      float v = 0.0f;
      if (bn_ok && (k0 + k) < K) v = B[(size_t)(k0 + k) * N + bn_g];
      ldsB[bn_c * LDS_STRIDE + k] = (h16)v;
    }
    if (bn_ok && k0 + 32 < K)
      __builtin_prefetch(&B[(size_t)(k0 + 32) * N + bn_g], 0, 1); // global_prefetch_b8
    __syncthreads();

    // ---- fragments via 16B LDS loads ----
    const v8h* pa0 = (const v8h*)&ldsA[l15 * LDS_STRIDE + hi * 8];
    const v8h* pa1 = (const v8h*)&ldsA[l15 * LDS_STRIDE + hi * 8 + 16];
    const int  bn_l = waveId * 16 + l15;
    const v8h* pb0 = (const v8h*)&ldsB[bn_l * LDS_STRIDE + hi * 8];
    const v8h* pb1 = (const v8h*)&ldsB[bn_l * LDS_STRIDE + hi * 8 + 16];
    v16h a = cat8(pa0[0], pa1[0]);
    v16h b = cat8(pb0[0], pb1[0]);

    acc = __builtin_amdgcn_wmma_f32_16x16x32_f16(false, a, false, b,
                                                 (short)0, acc, false, false);
    __syncthreads();
  }

  const int n = nBase + waveId * 16 + l15;
#pragma unroll
  for (int r = 0; r < 8; ++r) {
    int m = mBase + r + hi * 8;
    if (m < M && n < N) {
      float v = acc[r] + bias[n];
      if (relu) v = v > 0.0f ? v : 0.0f;
      if (outH) outH[(size_t)m * N + n] = (h16)v;
      if (outF) outF[(size_t)m * N + n] = v;
    }
  }
}

// ---------------------------------------------------------------------------
// 2x2 max pool (stride 2) on NCHW f16
// ---------------------------------------------------------------------------
__global__ void k_maxpool2(const h16* __restrict__ in, h16* __restrict__ out,
                           int C, int H, int W) {
  int Ho = H / 2, Wo = W / 2;
  int tot = C * Ho * Wo;
  int i = blockIdx.x * blockDim.x + threadIdx.x;
  if (i >= tot) return;
  int c = i / (Ho * Wo);
  int r = i % (Ho * Wo);
  int y = r / Wo, x = r % Wo;
  const h16* p = in + (size_t)c * H * W + (size_t)(2 * y) * W + 2 * x;
  float m = fmaxf(fmaxf((float)p[0], (float)p[1]),
                  fmaxf((float)p[W], (float)p[W + 1]));
  out[i] = (h16)m;
}

// ---------------------------------------------------------------------------
// RPN reshape: conv outputs (NCHW) -> (HW*9, 4)/(HW*9, 2) f32 + fg softmax
// ---------------------------------------------------------------------------
__global__ void k_rpn_reshape(const h16* __restrict__ locs, const h16* __restrict__ scores,
                              float* __restrict__ o_locs, float* __restrict__ o_scores,
                              float* __restrict__ fg) {
  int i = blockIdx.x * blockDim.x + threadIdx.x;
  if (i >= N_ANCH) return;
  int a = i % 9;
  int p = i / 9;
  int y = p / FWw, x = p % FWw;
  int sp = y * FWw + x;
#pragma unroll
  for (int c = 0; c < 4; ++c)
    o_locs[i * 4 + c] = (float)locs[(size_t)(a * 4 + c) * (FHh * FWw) + sp];
  float s0 = (float)scores[(size_t)(a * 2 + 0) * (FHh * FWw) + sp];
  float s1 = (float)scores[(size_t)(a * 2 + 1) * (FHh * FWw) + sp];
  o_scores[i * 2 + 0] = s0;
  o_scores[i * 2 + 1] = s1;
  fg[i] = 1.0f / (1.0f + expf(s0 - s1));   // softmax channel 1
}

// ---------------------------------------------------------------------------
// Anchor generation
// ---------------------------------------------------------------------------
__global__ void k_anchors(float* __restrict__ anc) {
  int i = blockIdx.x * blockDim.x + threadIdx.x;
  if (i >= N_ANCH) return;
  const float ratios[3] = {0.5f, 1.0f, 2.0f};
  const float scales[3] = {8.0f, 16.0f, 32.0f};
  int a = i % 9;
  int p = i / 9;
  int y = p / FWw, x = p % FWw;
  int ri = a / 3, si = a % 3;
  float h = 16.0f * scales[si] * sqrtf(ratios[ri]);
  float w = 16.0f * scales[si] * sqrtf(1.0f / ratios[ri]);
  float sy = y * 16.0f, sx = x * 16.0f;
  anc[i * 4 + 0] = sy + 8.0f - 0.5f * h;
  anc[i * 4 + 1] = sx + 8.0f - 0.5f * w;
  anc[i * 4 + 2] = sy + 8.0f + 0.5f * h;
  anc[i * 4 + 3] = sx + 8.0f + 0.5f * w;
}

// ---------------------------------------------------------------------------
// Proposal decode + clip + validity + sort-key setup (pads to SORT_N)
// ---------------------------------------------------------------------------
__global__ void k_proposals(const float* __restrict__ anc, const float* __restrict__ loc,
                            const float* __restrict__ fg, float* __restrict__ rois_all,
                            int* __restrict__ valid, float* __restrict__ keys,
                            int* __restrict__ vals) {
  int i = blockIdx.x * blockDim.x + threadIdx.x;
  if (i >= SORT_N) return;
  if (i >= N_ANCH) { keys[i] = -3.0e38f; vals[i] = 0; return; }
  float ay1 = anc[i*4+0], ax1 = anc[i*4+1], ay2 = anc[i*4+2], ax2 = anc[i*4+3];
  float h = ay2 - ay1, w = ax2 - ax1;
  float cy = ay1 + 0.5f * h, cx = ax1 + 0.5f * w;
  float dy = loc[i*4+0], dx = loc[i*4+1], dh = loc[i*4+2], dw = loc[i*4+3];
  float cy2 = dy * h + cy, cx2 = dx * w + cx;
  float h2 = expf(dh) * h, w2 = expf(dw) * w;
  float y1 = cy2 - 0.5f * h2, x1 = cx2 - 0.5f * w2;
  float y2 = cy2 + 0.5f * h2, x2 = cx2 + 0.5f * w2;
  y1 = fminf(fmaxf(y1, 0.0f), (float)H_IMG);
  y2 = fminf(fmaxf(y2, 0.0f), (float)H_IMG);
  x1 = fminf(fmaxf(x1, 0.0f), (float)W_IMG);
  x2 = fminf(fmaxf(x2, 0.0f), (float)W_IMG);
  rois_all[i*4+0] = y1; rois_all[i*4+1] = x1;
  rois_all[i*4+2] = y2; rois_all[i*4+3] = x2;
  int v = (y2 - y1 >= 16.0f) && (x2 - x1 >= 16.0f);
  valid[i] = v;
  keys[i]  = v ? fg[i] : NEGV;
  vals[i]  = i;
}

// ---------------------------------------------------------------------------
// Bitonic sort stage (descending by key), key/value pairs in global memory
// ---------------------------------------------------------------------------
__global__ void k_bitonic(float* __restrict__ keys, int* __restrict__ vals, int j, int k) {
  int i = blockIdx.x * blockDim.x + threadIdx.x;
  int ixj = i ^ j;
  if (i < SORT_N && ixj > i && ixj < SORT_N) {
    bool up = ((i & k) == 0);            // up => larger element first (descending)
    float ki = keys[i], kj = keys[ixj];
    bool sw = up ? (ki < kj) : (ki > kj);
    if (sw) {
      keys[i] = kj; keys[ixj] = ki;
      int t = vals[i]; vals[i] = vals[ixj]; vals[ixj] = t;
    }
  }
}

// ---------------------------------------------------------------------------
// Gather top N_PRE sorted rois into SoA + initial suppression (= !valid)
// ---------------------------------------------------------------------------
__global__ void k_gather_sorted(const int* __restrict__ vals, const float* __restrict__ rois_all,
                                const int* __restrict__ valid,
                                float* __restrict__ y1, float* __restrict__ x1,
                                float* __restrict__ y2, float* __restrict__ x2,
                                float* __restrict__ area, int* __restrict__ supp) {
  int i = blockIdx.x * blockDim.x + threadIdx.x;
  if (i >= N_PRE) return;
  int idx = vals[i];
  float a = rois_all[idx*4+0], b = rois_all[idx*4+1];
  float c = rois_all[idx*4+2], d = rois_all[idx*4+3];
  y1[i] = a; x1[i] = b; y2[i] = c; x2[i] = d;
  area[i] = (c - a) * (d - b);
  supp[i] = valid[idx] ? 0 : 1;
}

// ---------------------------------------------------------------------------
// Sequential-greedy NMS, one workgroup; suppression flags in LDS
// ---------------------------------------------------------------------------
__global__ __launch_bounds__(256)
void k_nms(const float* __restrict__ y1, const float* __restrict__ x1,
           const float* __restrict__ y2, const float* __restrict__ x2,
           const float* __restrict__ area, int* __restrict__ supp) {
  __shared__ unsigned char s[N_PRE];
  for (int i = threadIdx.x; i < N_PRE; i += 256) s[i] = (unsigned char)supp[i];
  __syncthreads();
  for (int i = 0; i < N_PRE; ++i) {
    if (!s[i]) {
      float Y1 = y1[i], X1 = x1[i], Y2 = y2[i], X2 = x2[i], A = area[i];
      for (int j = i + 1 + threadIdx.x; j < N_PRE; j += 256) {
        float yy1 = fmaxf(Y1, y1[j]), xx1 = fmaxf(X1, x1[j]);
        float yy2 = fminf(Y2, y2[j]), xx2 = fminf(X2, x2[j]);
        float inter = fmaxf(yy2 - yy1, 0.0f) * fmaxf(xx2 - xx1, 0.0f);
        float un = A + area[j] - inter;
        float iou = un > 0.0f ? inter / un : 0.0f;
        if (iou > NMS_TH) s[j] = 1;
      }
    }
    __syncthreads();
  }
  for (int i = threadIdx.x; i < N_PRE; i += 256) supp[i] = s[i];
}

// ---------------------------------------------------------------------------
// Select first N_POST kept rois (sorted order); write rois, ok mask, zero idx
// ---------------------------------------------------------------------------
__global__ void k_select(const int* __restrict__ supp,
                         const float* __restrict__ y1, const float* __restrict__ x1,
                         const float* __restrict__ y2, const float* __restrict__ x2,
                         float* __restrict__ o_rois, float* __restrict__ o_ridx,
                         int* __restrict__ okf) {
  if (threadIdx.x != 0 || blockIdx.x != 0) return;
  int outp = 0, cnt = 0;
  for (int i = 0; i < N_PRE; ++i) {
    if (!supp[i]) {
      if (outp < N_POST) {
        o_rois[outp*4+0] = y1[i]; o_rois[outp*4+1] = x1[i];
        o_rois[outp*4+2] = y2[i]; o_rois[outp*4+3] = x2[i];
        ++outp;
      }
      ++cnt;
    }
  }
  for (int i = 0; i < N_POST; ++i) {
    okf[i] = (i < cnt) ? 1 : 0;
    if (i >= outp) {
      o_rois[i*4+0] = 0.f; o_rois[i*4+1] = 0.f;
      o_rois[i*4+2] = 0.f; o_rois[i*4+3] = 0.f;
    }
    o_ridx[i] = 0.0f;
  }
}

// ---------------------------------------------------------------------------
// ROI max pooling 7x7 (window capped at 8x8 like reference), output f16 row
// ---------------------------------------------------------------------------
__global__ __launch_bounds__(256)
void k_roipool(const h16* __restrict__ feat, const float* __restrict__ rois,
               const int* __restrict__ okf, h16* __restrict__ pooled) {
  int r = blockIdx.x;
  float r0 = rois[r*4+0], r1 = rois[r*4+1], r2 = rois[r*4+2], r3 = rois[r*4+3];
  int ok = okf[r];
  int sy = (int)floorf(r0 * 0.0625f + 0.5f);
  int sx = (int)floorf(r1 * 0.0625f + 0.5f);
  int ey = (int)floorf(r2 * 0.0625f + 0.5f);
  int ex = (int)floorf(r3 * 0.0625f + 0.5f);
  float bh = (float)imax(ey - sy + 1, 1) / 7.0f;
  float bw = (float)imax(ex - sx + 1, 1) / 7.0f;
  for (int o = threadIdx.x; o < 512 * 49; o += 256) {
    int c = o / 49;
    int rem = o % 49;
    int ph = rem / 7, pw = rem % 7;
    int hs = imin(imax((int)floorf(ph * bh) + sy, 0), FHh);
    int he = imin(imax((int)ceilf((ph + 1) * bh) + sy, 0), FHh);
    int ws = imin(imax((int)floorf(pw * bw) + sx, 0), FWw);
    int we = imin(imax((int)ceilf((pw + 1) * bw) + sx, 0), FWw);
    bool empty = (he <= hs) || (we <= ws);
    int he2 = imin(he, hs + 8);          // reference iterates k in [0,8)
    int we2 = imin(we, ws + 8);
    float m = NEGV;
    for (int y = hs; y < he2; ++y)
      for (int x = ws; x < we2; ++x)
        m = fmaxf(m, (float)feat[(size_t)c * (FHh * FWw) + y * FWw + x]);
    float v = empty ? 0.0f : m;
    if (!ok) v = 0.0f;
    pooled[(size_t)r * 25088 + o] = (h16)v;
  }
}

// ===========================================================================
extern "C" void kernel_launch(void* const* d_in, const int* in_sizes, int n_in,
                              void* d_out, int out_size, void* d_ws, size_t ws_size,
                              hipStream_t stream) {
  (void)in_sizes; (void)n_in; (void)out_size; (void)ws_size;

  // ---- input pointers (setup_inputs dict order, tuples flattened) ----
  const float* x = (const float*)d_in[0];
  // vgg: 13 (w,b) pairs at d_in[1..26]
  const float* wc  = (const float*)d_in[27];
  const float* bc  = (const float*)d_in[28];
  const float* wl  = (const float*)d_in[29];
  const float* bl  = (const float*)d_in[30];
  const float* wsc = (const float*)d_in[31];
  const float* bsc = (const float*)d_in[32];
  const float* w1  = (const float*)d_in[33];
  const float* b1  = (const float*)d_in[34];
  const float* w2  = (const float*)d_in[35];
  const float* b2  = (const float*)d_in[36];
  const float* wcl = (const float*)d_in[37];
  const float* bcl = (const float*)d_in[38];
  const float* wsh = (const float*)d_in[39];
  const float* bsh = (const float*)d_in[40];

  // ---- output regions (flat f32) ----
  float* out     = (float*)d_out;
  float* o_cls   = out;               // 300*84
  float* o_sc    = out + 25200;       // 300*21
  float* o_rois  = out + 31500;       // 300*4
  float* o_ridx  = out + 32700;       // 300
  float* o_rlocs = out + 33000;       // 6912*4
  float* o_rsc   = out + 60648;       // 6912*2
  float* o_anc   = out + 74472;       // 6912*4

  // ---- workspace bump allocator (~70 MB total) ----
  char* wsb = (char*)d_ws;
  size_t off = 0;
  auto alloc = [&](size_t bytes) -> void* {
    off = (off + 255) & ~(size_t)255;
    void* p = wsb + off;
    off += bytes;
    return p;
  };
  const size_t MAXACT = (size_t)64 * H_IMG * W_IMG;   // largest activation (elems)
  h16*   bufA   = (h16*)alloc(MAXACT * sizeof(h16));
  h16*   bufB   = (h16*)alloc(MAXACT * sizeof(h16));
  h16*   locsb  = (h16*)alloc((size_t)36 * FHh * FWw * sizeof(h16));
  h16*   scob   = (h16*)alloc((size_t)18 * FHh * FWw * sizeof(h16));
  float* fg     = (float*)alloc((size_t)N_ANCH * 4);
  float* roisA  = (float*)alloc((size_t)N_ANCH * 4 * 4);
  int*   valid  = (int*)alloc((size_t)N_ANCH * 4);
  float* keys   = (float*)alloc((size_t)SORT_N * 4);
  int*   vals   = (int*)alloc((size_t)SORT_N * 4);
  float* sy1    = (float*)alloc((size_t)N_PRE * 4);
  float* sx1    = (float*)alloc((size_t)N_PRE * 4);
  float* sy2    = (float*)alloc((size_t)N_PRE * 4);
  float* sx2    = (float*)alloc((size_t)N_PRE * 4);
  float* sarea  = (float*)alloc((size_t)N_PRE * 4);
  int*   supp   = (int*)alloc((size_t)N_PRE * 4);
  int*   okf    = (int*)alloc((size_t)N_POST * 4);
  h16*   pooled = (h16*)alloc((size_t)N_POST * 25088 * sizeof(h16));
  h16*   fc1o   = (h16*)alloc((size_t)N_POST * 4096 * sizeof(h16));
  h16*   fc2o   = (h16*)alloc((size_t)N_POST * 4096 * sizeof(h16));

  // ---- 1. image f32 -> f16 ----
  {
    int n = 3 * H_IMG * W_IMG;
    k_cvt_f16<<<(n + 255) / 256, 256, 0, stream>>>(x, bufA, n);
  }

  // ---- 2. VGG-16 extractor (WMMA implicit-GEMM convs + maxpools) ----
  const int cfg[17] = {64,64,-1,128,128,-1,256,256,256,-1,512,512,512,-1,512,512,512};
  h16* cur = bufA;
  h16* nxt = bufB;
  int cin = 3, H = H_IMG, Wd = W_IMG, convIdx = 0;
  for (int s = 0; s < 17; ++s) {
    if (cfg[s] < 0) {
      int Ho = H / 2, Wo = Wd / 2, tot = cin * Ho * Wo;
      k_maxpool2<<<(tot + 255) / 256, 256, 0, stream>>>(cur, nxt, cin, H, Wd);
      H = Ho; Wd = Wo;
    } else {
      int cout = cfg[s];
      const float* Wp = (const float*)d_in[1 + 2 * convIdx];
      const float* Bp = (const float*)d_in[2 + 2 * convIdx];
      ++convIdx;
      int HWt = H * Wd;
      dim3 g((HWt + 15) / 16, (cout + 15) / 16);
      k_conv_wmma<<<g, 32, 0, stream>>>(Wp, Bp, cur, nxt, cin, cout, H, Wd, 3, 3, 1, 1);
      cin = cout;
    }
    h16* t = cur; cur = nxt; nxt = t;
  }
  h16* feat = cur;   // 512 x 24 x 32 (f16)
  h16* rpnh = nxt;

  // ---- 3. RPN convs ----
  k_conv_wmma<<<dim3(48, 32), 32, 0, stream>>>(wc,  bc,  feat, rpnh,  512, 512, FHh, FWw, 3, 3, 1, 1);
  k_conv_wmma<<<dim3(48, 3),  32, 0, stream>>>(wl,  bl,  rpnh, locsb, 512, 36,  FHh, FWw, 1, 1, 0, 0);
  k_conv_wmma<<<dim3(48, 2),  32, 0, stream>>>(wsc, bsc, rpnh, scob,  512, 18,  FHh, FWw, 1, 1, 0, 0);

  // ---- 4. reshape / softmax fg, anchors, proposal decode ----
  k_rpn_reshape<<<(N_ANCH + 255) / 256, 256, 0, stream>>>(locsb, scob, o_rlocs, o_rsc, fg);
  k_anchors<<<(N_ANCH + 255) / 256, 256, 0, stream>>>(o_anc);
  k_proposals<<<SORT_N / 256, 256, 0, stream>>>(o_anc, o_rlocs, fg, roisA, valid, keys, vals);

  // ---- 5. bitonic argsort (descending score) ----
  for (int k = 2; k <= SORT_N; k <<= 1)
    for (int j = k >> 1; j > 0; j >>= 1)
      k_bitonic<<<SORT_N / 256, 256, 0, stream>>>(keys, vals, j, k);

  // ---- 6. NMS + keep-selection ----
  k_gather_sorted<<<(N_PRE + 255) / 256, 256, 0, stream>>>(vals, roisA, valid,
                                                           sy1, sx1, sy2, sx2, sarea, supp);
  k_nms<<<1, 256, 0, stream>>>(sy1, sx1, sy2, sx2, sarea, supp);
  k_select<<<1, 1, 0, stream>>>(supp, sy1, sx1, sy2, sx2, o_rois, o_ridx, okf);

  // ---- 7. ROI max-pool 7x7 -> [300, 25088] f16 ----
  k_roipool<<<N_POST, 256, 0, stream>>>(feat, o_rois, okf, pooled);

  // ---- 8. FC head (LDS-staged WMMA GEMMs, 8 waves/workgroup) ----
  k_gemm_wmma<<<dim3((4096 + GT_N - 1) / GT_N, (N_POST + 15) / 16), 256, 0, stream>>>(
      pooled, w1, b1, N_POST, 4096, 25088, 1, fc1o, (float*)nullptr);
  k_gemm_wmma<<<dim3((4096 + GT_N - 1) / GT_N, (N_POST + 15) / 16), 256, 0, stream>>>(
      fc1o, w2, b2, N_POST, 4096, 4096, 1, fc2o, (float*)nullptr);
  k_gemm_wmma<<<dim3((84 + GT_N - 1) / GT_N, (N_POST + 15) / 16), 256, 0, stream>>>(
      fc2o, wcl, bcl, N_POST, 84, 4096, 0, (h16*)nullptr, o_cls);
  k_gemm_wmma<<<dim3((21 + GT_N - 1) / GT_N, (N_POST + 15) / 16), 256, 0, stream>>>(
      fc2o, wsh, bsh, N_POST, 21, 4096, 0, (h16*)nullptr, o_sc);
}
